// EDA_28020366639751
// MI455X (gfx1250) — matrix operand order repaired and checked by reading, not compiled, not run on hardware
//
#include <hip/hip_runtime.h>

// MI455X / gfx1250, wave32. Fused unrolled-softmax-unmixing kernel.
//   refactor: resid@D = t - alpha@G with G = D^T D precomputed (7x FLOP cut)
//   t (K=224) : split-bf16 v_wmma_f32_16x16x32_bf16 (xh*Dh + xl*Dh + xh*Dl)
//   loop      : f32 v_wmma_f32_16x16x4_f32 on alpha@G (numerically sensitive)
//   x_hat     : bf16 WMMA
//   x staging : CDNA5 async global->LDS (ASYNCcnt), no VGPR round trip
// alpha/t stay in registers; x read once; outputs written once (memory-bound,
// floor ~= 0.5GB / 23.3TB/s ~= 22us).

typedef __attribute__((ext_vector_type(2)))  float  v2f;
typedef __attribute__((ext_vector_type(8)))  float  v8f;
typedef __attribute__((ext_vector_type(16))) __bf16 v16bf;

#define N_BANDS 224
#define N_END   32
#define PIX_PER_WAVE 16
#define WAVES_PER_BLOCK 8
#define PIX_PER_BLOCK 128

union BF16x16 { v16bf v; unsigned u[8]; };

// Native bf16 converts (gfx1250 has hw bf16; fptrunc = RNE, matches ref rounding)
__device__ __forceinline__ unsigned pkbf(float a, float b) {
  unsigned short l = __builtin_bit_cast(unsigned short, (__bf16)a);
  unsigned short h = __builtin_bit_cast(unsigned short, (__bf16)b);
  return (unsigned)l | ((unsigned)h << 16);
}
__device__ __forceinline__ float bfhi(float a) { return (float)(__bf16)a; }

// 32-bit LDS byte offset of a generic pointer to __shared__ (generic->as3 cast)
typedef __attribute__((address_space(3))) void lds_void;
__device__ __forceinline__ unsigned lds_off(void* p) {
  return (unsigned)(unsigned long long)(lds_void*)(p);
}

// LDS is in-order per wave; these are compiler fences + explicit CDNA5
// split-counter drains for cross-lane / cross-pipe LDS dependencies.
#define LDS_FENCE()   __asm__ volatile("s_wait_dscnt 0x0" ::: "memory")
#define ASYNC_FENCE() __asm__ volatile("s_wait_asynccnt 0x0" ::: "memory")

// ---------------------------------------------------------------------------
// K1: G = D^T @ D  (32x32 f32) into d_ws. One block; D staged in LDS.
// ---------------------------------------------------------------------------
__global__ void EDA_gram_kernel(const float* __restrict__ D, float* __restrict__ G) {
  __shared__ float Ds[N_BANDS * N_END];
  for (int i = threadIdx.x; i < N_BANDS * N_END; i += blockDim.x) Ds[i] = D[i];
  __syncthreads();
  for (int o = threadIdx.x; o < N_END * N_END; o += blockDim.x) {
    int i = o >> 5, j = o & 31;
    float s = 0.f;
    for (int k = 0; k < N_BANDS; ++k) s = fmaf(Ds[k * 32 + i], Ds[k * 32 + j], s);
    G[o] = s;
  }
}

// ---------------------------------------------------------------------------
// K2: fused pipeline. 256 threads = 8 waves, 16 pixels per wave.
// ---------------------------------------------------------------------------
__global__ __launch_bounds__(256) void EDA_unmix_kernel(
    const float* __restrict__ x, const float* __restrict__ D,
    const float* __restrict__ etas, int nEta,
    const float* __restrict__ G,
    float* __restrict__ xhat, float* __restrict__ alphaOut)
{
  // Per-wave scratch (576 f32): x-chunk staging [16 x 36] OR transpose buf
  // [16 x 34] (disjoint lifetimes, strides padded for bank-conflict-free use).
  __shared__ float    Ubuf[WAVES_PER_BLOCK * 576];        // 18432 B
  __shared__ unsigned Dth[32 * 113];                      // D^T bf16 hi, stride 226 bf16
  __shared__ unsigned Dtl[32 * 113];                      // D^T bf16 lo (split residual)
  __shared__ unsigned Dbf[N_BANDS * 17];                  // D bf16 row-major, stride 34 bf16
                                                          // total 62.6 KB LDS

  const int tid  = threadIdx.x;
  const int wv   = tid >> 5;
  const int lane = tid & 31;
  const int r    = lane & 15;   // row within 16 (pixel for A, N-col for B/C/D)
  const int h    = lane >> 4;   // lane half selector in WMMA layouts

  // ---- block-wide staging of D ----
  for (int idx = tid; idx < 32 * 112; idx += 256) {       // D^T hi/lo (band pairs)
    int e = idx / 112, bp = idx - e * 112;
    float a = D[(2 * bp    ) * 32 + e];
    float b = D[(2 * bp + 1) * 32 + e];
    Dth[e * 113 + bp] = pkbf(a, b);
    Dtl[e * 113 + bp] = pkbf(a - bfhi(a), b - bfhi(b));
  }
  for (int idx = tid; idx < N_BANDS * 16; idx += 256) {   // D bf16 (endmember pairs)
    int band = idx >> 4, ep = idx & 15;
    v2f d = *(const v2f*)&D[band * 32 + 2 * ep];
    Dbf[band * 17 + ep] = pkbf(d[0], d[1]);
  }
  __syncthreads();

  float* myU = &Ubuf[wv * 576];
  const unsigned uBase = lds_off(myU);
  const long pixW = (long)blockIdx.x * PIX_PER_BLOCK + wv * PIX_PER_WAVE;

  // ======== Stage 1: t = x @ D, split-bf16, K=224 in 7 chunks of 32 ========
  v8f tLo = {}, tHi = {};   // C/D layout: element (m = v + 8h, n = r)
  #pragma unroll
  for (int c = 0; c < 7; ++c) {
    const int kc = c * 32;
    LDS_FENCE();                                   // prior DS reads of myU done
    // async global->LDS staging of the 16x32 f32 chunk (row stride 36 f32);
    // per lane: 16B from global to 16B-aligned LDS offset, ASYNCcnt-tracked.
    #pragma unroll
    for (int i = 0; i < 4; ++i) {
      int flat = i * 128 + lane * 4;
      int row = flat >> 5, col = flat & 31;
      unsigned long long ga =
          (unsigned long long)(const void*)&x[(pixW + row) * N_BANDS + kc + col];
      unsigned la = uBase + (unsigned)((row * 36 + col) * 4);
      __asm__ volatile("global_load_async_to_lds_b128 %0, %1, off"
                       :: "v"(la), "v"(ga) : "memory");
    }
    ASYNC_FENCE();                                 // LDS writes visible to DS pipe
    BF16x16 Ah, Al;                                // bf16 A layout 16x32
    #pragma unroll
    for (int v = 0; v < 8; ++v) {
      int k = (v >> 2) * 16 + (v & 3) * 2 + 8 * h; // K pair base for reg v
      v2f p = *(const v2f*)&myU[r * 36 + k];
      Ah.u[v] = pkbf(p[0], p[1]);
      Al.u[v] = pkbf(p[0] - bfhi(p[0]), p[1] - bfhi(p[1]));
    }
    BF16x16 BhLo, BhHi, BlLo, BlHi;                // bf16 B layout 32x16 (from D^T)
    #pragma unroll
    for (int v = 0; v < 8; ++v) {
      int wd = (kc >> 1) + (v >> 2) * 8 + 4 * h + (v & 3);
      BhLo.u[v] = Dth[(     r) * 113 + wd];
      BhHi.u[v] = Dth[(16 + r) * 113 + wd];
      BlLo.u[v] = Dtl[(     r) * 113 + wd];
      BlHi.u[v] = Dtl[(16 + r) * 113 + wd];
    }
    tLo = __builtin_amdgcn_wmma_f32_16x16x32_bf16(false, Ah.v, false, BhLo.v, (short)0, tLo, false, false);
    tLo = __builtin_amdgcn_wmma_f32_16x16x32_bf16(false, Al.v, false, BhLo.v, (short)0, tLo, false, false);
    tLo = __builtin_amdgcn_wmma_f32_16x16x32_bf16(false, Ah.v, false, BlLo.v, (short)0, tLo, false, false);
    tHi = __builtin_amdgcn_wmma_f32_16x16x32_bf16(false, Ah.v, false, BhHi.v, (short)0, tHi, false, false);
    tHi = __builtin_amdgcn_wmma_f32_16x16x32_bf16(false, Al.v, false, BhHi.v, (short)0, tHi, false, false);
    tHi = __builtin_amdgcn_wmma_f32_16x16x32_bf16(false, Ah.v, false, BlHi.v, (short)0, tHi, false, false);
  }

  // ======== alpha0 = softmax(t): transpose C/D -> f32 A layout, lane-local rows ====
  LDS_FENCE();
  #pragma unroll
  for (int v = 0; v < 8; ++v) {
    myU[(v + 8 * h) * 34 + r]      = tLo[v];
    myU[(v + 8 * h) * 34 + 16 + r] = tHi[v];
  }
  LDS_FENCE();
  float alphaA[16];   // f32 A layout: alphaA[2j+i] = alpha[r][4j + 2h + i]
  {
    float tA[16];
    #pragma unroll
    for (int j = 0; j < 8; ++j) {
      v2f p = *(const v2f*)&myU[r * 34 + 4 * j + 2 * h];
      tA[2 * j] = p[0]; tA[2 * j + 1] = p[1];
    }
    float mx = tA[0];
    #pragma unroll
    for (int i = 1; i < 16; ++i) mx = fmaxf(mx, tA[i]);
    mx = fmaxf(mx, __shfl_xor(mx, 16, 32));        // row halves live in lane & lane^16
    float sum = 0.f;
    #pragma unroll
    for (int i = 0; i < 16; ++i) { float e = __expf(tA[i] - mx); alphaA[i] = e; sum += e; }
    sum += __shfl_xor(sum, 16, 32);
    float inv = 1.f / sum;
    #pragma unroll
    for (int i = 0; i < 16; ++i) alphaA[i] *= inv;
  }

  // ======== preload G B-operands once (G symmetric -> contiguous b64 pairs) ====
  v2f gB[2][8];   // f32 B layout 4x16 slices: gB[tile][j] regs (k=4j+2h, +1)
  #pragma unroll
  for (int tt = 0; tt < 2; ++tt)
    #pragma unroll
    for (int j = 0; j < 8; ++j)
      gB[tt][j] = *(const v2f*)&G[(16 * tt + r) * 32 + 4 * j + 2 * h];

  // ======== iteration loop: b = eta*(t - alpha@G); alpha <- softmax-reweight ====
  for (int it = 0; it < nEta; ++it) {
    float eta = etas[it];
    v8f sLo = {}, sHi = {};
    #pragma unroll
    for (int j = 0; j < 8; ++j) {                  // alpha@G, f32 WMMA, K=32
      v2f a; a[0] = alphaA[2 * j]; a[1] = alphaA[2 * j + 1];
      sLo = __builtin_amdgcn_wmma_f32_16x16x4_f32(false, a, false, gB[0][j], (short)0, sLo, false, false);
      sHi = __builtin_amdgcn_wmma_f32_16x16x4_f32(false, a, false, gB[1][j], (short)0, sHi, false, false);
    }
    LDS_FENCE();
    #pragma unroll
    for (int v = 0; v < 8; ++v) {                  // b in C/D layout -> LDS transpose
      myU[(v + 8 * h) * 34 + r]      = eta * (tLo[v] - sLo[v]);
      myU[(v + 8 * h) * 34 + 16 + r] = eta * (tHi[v] - sHi[v]);
    }
    LDS_FENCE();
    float bA[16];
    #pragma unroll
    for (int j = 0; j < 8; ++j) {
      v2f p = *(const v2f*)&myU[r * 34 + 4 * j + 2 * h];
      bA[2 * j] = p[0]; bA[2 * j + 1] = p[1];
    }
    float mx = bA[0];
    #pragma unroll
    for (int i = 1; i < 16; ++i) mx = fmaxf(mx, bA[i]);
    mx = fmaxf(mx, __shfl_xor(mx, 16, 32));
    float sum = 0.f, w[16];
    #pragma unroll
    for (int i = 0; i < 16; ++i) { w[i] = alphaA[i] * __expf(bA[i] - mx); sum += w[i]; }
    sum += __shfl_xor(sum, 16, 32);
    float inv = 1.f / sum;
    #pragma unroll
    for (int i = 0; i < 16; ++i) alphaA[i] = w[i] * inv;
  }

  // ======== outputs ========
  #pragma unroll
  for (int j = 0; j < 8; ++j) {                    // alpha [N,32] f32, b64 stores
    v2f a; a[0] = alphaA[2 * j]; a[1] = alphaA[2 * j + 1];
    *(v2f*)&alphaOut[(pixW + r) * N_END + 4 * j + 2 * h] = a;
  }

  // x_hat = alpha @ D^T (bf16 WMMA): rebuild alpha in bf16 A layout via LDS
  LDS_FENCE();
  #pragma unroll
  for (int j = 0; j < 8; ++j) {
    v2f a; a[0] = alphaA[2 * j]; a[1] = alphaA[2 * j + 1];
    *(v2f*)&myU[r * 34 + 4 * j + 2 * h] = a;
  }
  LDS_FENCE();
  BF16x16 Abf;
  #pragma unroll
  for (int v = 0; v < 8; ++v) {
    int k = (v >> 2) * 16 + (v & 3) * 2 + 8 * h;
    v2f p = *(const v2f*)&myU[r * 34 + k];
    Abf.u[v] = pkbf(p[0], p[1]);
  }
  #pragma unroll
  for (int bt = 0; bt < 14; ++bt) {                // 14 band tiles of 16
    BF16x16 Bb;
    #pragma unroll
    for (int v = 0; v < 8; ++v)
      Bb.u[v] = Dbf[(16 * bt + r) * 17 + (v >> 2) * 8 + 4 * h + (v & 3)];
    v8f xh = {};
    xh = __builtin_amdgcn_wmma_f32_16x16x32_bf16(false, Abf.v, false, Bb.v, (short)0, xh, false, false);
    #pragma unroll
    for (int v = 0; v < 8; ++v)
      xhat[(pixW + v + 8 * h) * N_BANDS + 16 * bt + r] = xh[v];
  }
}

// ---------------------------------------------------------------------------
extern "C" void kernel_launch(void* const* d_in, const int* in_sizes, int n_in,
                              void* d_out, int out_size, void* d_ws, size_t ws_size,
                              hipStream_t stream) {
  const float* x    = (const float*)d_in[0];
  const float* D    = (const float*)d_in[1];
  const float* etas = (const float*)d_in[2];
  const int nEta = in_sizes[2];
  const long N = (long)in_sizes[0] / N_BANDS;

  float* G        = (float*)d_ws;            // 32*32*4 = 4 KB scratch
  float* xhat     = (float*)d_out;           // [N, 224]
  float* alphaOut = xhat + N * N_BANDS;      // [N, 32]

  hipLaunchKernelGGL(EDA_gram_kernel, dim3(1), dim3(256), 0, stream, D, G);
  hipLaunchKernelGGL(EDA_unmix_kernel, dim3((unsigned)(N / PIX_PER_BLOCK)), dim3(256), 0, stream,
                     x, D, etas, nEta, G, xhat, alphaOut);
}
